// GrapherModule_35433480192925
// MI455X (gfx1250) — matrix operand order repaired and checked by reading, not compiled, not run on hardware
//
#include <hip/hip_runtime.h>
#include <hip/hip_bf16.h>

// ---------------- problem constants ----------------
#define BB   128
#define NN   400
#define FF   128
#define KNN  5
#define NODES (BB * NN)          // 51200
#define ADJW  16                 // u32 words per adjacency row (400 bits -> 13, pad 16)

// ---------------- WMMA types ----------------
typedef __attribute__((ext_vector_type(16))) __bf16 v16bf;
typedef __attribute__((ext_vector_type(8)))  float  v8f;
// TDM descriptor vector types
typedef __attribute__((ext_vector_type(4))) unsigned v4u;
typedef __attribute__((ext_vector_type(8))) int      v8i_;
typedef __attribute__((ext_vector_type(4))) int      v4i_;

union V16 { v16bf v; unsigned short u[16]; unsigned d[8]; };

// ---------------- helpers ----------------
__device__ inline unsigned f2bf(float f) {            // f32 -> bf16 bits (RNE)
  unsigned u = __builtin_bit_cast(unsigned, f);
  return (u + 0x7FFFu + ((u >> 16) & 1u)) >> 16;
}
__device__ inline float bf2f(unsigned s) {            // bf16 bits -> f32
  return __builtin_bit_cast(float, s << 16);
}
__device__ inline unsigned short lds_u16(const unsigned* row, int c) {
  unsigned v = row[c >> 1];
  return (unsigned short)((c & 1) ? (v >> 16) : (v & 0xffffu));
}

// ---------------- kernel 0: zero adjacency ----------------
__global__ void zero_kernel(unsigned* __restrict__ p, int n) {
  int i = blockIdx.x * blockDim.x + threadIdx.x;
  if (i < n) p[i] = 0u;
}

// ---------------- kernel 1: x f32 -> bf16 into Acat1 right half ----------------
// Acat layout (u32 view): [node][128] u32 = 256 bf16. cols 0..127 = m, 128..255 = h.
__global__ void cvt_x_kernel(const float* __restrict__ x, unsigned* __restrict__ Acat1) {
  int p = blockIdx.x * blockDim.x + threadIdx.x;     // 0 .. NODES*64
  if (p >= NODES * 64) return;
  int node = p >> 6, cp = p & 63;
  float f0 = x[node * FF + cp * 2 + 0];
  float f1 = x[node * FF + cp * 2 + 1];
  Acat1[node * 128 + 64 + cp] = f2bf(f0) | (f2bf(f1) << 16);
}

// ---------------- kernel 2: weight prep (fragment-ready layout) ----------------
// For col-tile nt(0..7), k-step kt(0..7), lane l(0..31), elem e(0..15):
//   WcatF[(((nt*8+kt)*32)+l)*16 + e] = bf16( Wcat[k][n] ),
//   k = kt*32 + (e>>3)*16 + (l>>4)*8 + (e&7),  n = nt*16 + (l&15),
//   Wcat[k][n] = k<128 ? Wl[n][k] : Wr[n][k-128].
// B-fragment load in GEMM = 32 contiguous bytes per lane (2 x b128).
__global__ void prep_w_kernel(const float* __restrict__ W1l, const float* __restrict__ W1r,
                              const float* __restrict__ W2l, const float* __restrict__ W2r,
                              unsigned short* __restrict__ WcatF1, unsigned short* __restrict__ WcatF2) {
  int tid = blockIdx.x * blockDim.x + threadIdx.x;   // 0 .. 65535
  int which = tid >> 15;
  int r = tid & 32767;
  int e  = r & 15;
  int l  = (r >> 4) & 31;
  int kt = (r >> 9) & 7;
  int nt = (r >> 12) & 7;
  int k = kt * 32 + ((e >> 3) << 4) + ((l >> 4) << 3) + (e & 7);
  int n = nt * 16 + (l & 15);
  const float* Wl = which ? W2l : W1l;
  const float* Wr = which ? W2r : W1r;
  float v = (k < 128) ? Wl[n * 128 + k] : Wr[n * 128 + (k - 128)];
  (which ? WcatF2 : WcatF1)[r] = (unsigned short)f2bf(v);
}

// ---------------- kernel 3: Gram (WMMA bf16) + top-5 + symmetrized adjacency ----------------
// One wave per (batch, 16-row tile). grid = BB*25, block = 32.
__global__ void __launch_bounds__(32)
gram_topk_kernel(const unsigned* __restrict__ xbf /* Acat1 u32 view */, unsigned* __restrict__ adj) {
  __shared__ unsigned xi[16][64];    // 16 rows x 128 bf16
  __shared__ unsigned xj[16][64];
  __shared__ float sqj[16];
  __shared__ float d2[16][NN];       // 25.6 KB

  int blk = blockIdx.x;
  int b = blk / 25, rt = blk % 25;
  int rowbase = rt * 16;
  int nodebase = b * NN + rowbase;
  int l = threadIdx.x;
  int half = l >> 4, lm = l & 15;

  for (int idx = l; idx < 16 * 64; idx += 32) {
    int r = idx >> 6, c = idx & 63;
    xi[r][c] = xbf[(nodebase + r) * 128 + 64 + c];
  }
  __syncthreads();

  for (int jt = 0; jt < 25; ++jt) {
    int colnodebase = b * NN + jt * 16;
    for (int idx = l; idx < 16 * 64; idx += 32) {
      int r = idx >> 6, c = idx & 63;
      xj[r][c] = xbf[(colnodebase + r) * 128 + 64 + c];
    }
    __syncthreads();
    if (l < 16) {                    // column sq-norms (bf16-consistent)
      float s = 0.f;
      for (int c = 0; c < 64; ++c) {
        unsigned v = xj[l][c];
        float f0 = bf2f(v & 0xffffu), f1 = bf2f(v >> 16);
        s += f0 * f0 + f1 * f1;
      }
      sqj[l] = s;
    }
    __syncthreads();

    v8f acc = {};
    #pragma unroll
    for (int kt = 0; kt < 4; ++kt) {
      int k0 = kt * 32;
      V16 a, bm;
      #pragma unroll
      for (int e = 0; e < 16; ++e) {
        int kla = ((e >> 3) << 4) + (half << 3) + (e & 7);   // A layout
        a.u[e]  = lds_u16(xi[lm], k0 + kla);
        int klb = ((e >> 3) << 4) + (half << 3) + (e & 7);   // B layout
        bm.u[e] = lds_u16(xj[lm], k0 + klb);
      }
      acc = __builtin_amdgcn_wmma_f32_16x16x32_bf16(false, a.v, false, bm.v,
                                                    (short)0, acc, false, false);
    }
    // d2' = sq_j - 2*dot  (same ranking as full d2 for a fixed row)
    #pragma unroll
    for (int r = 0; r < 8; ++r) {
      int row = r + (half << 3);
      d2[row][jt * 16 + lm] = sqj[lm] - 2.f * acc[r];
    }
    __syncthreads();
  }

  if (l < 16) {                      // streaming top-5 per row
    float bv[KNN]; int bi[KNN];
    #pragma unroll
    for (int s = 0; s < KNN; ++s) { bv[s] = 3.4e38f; bi[s] = 0; }
    for (int j = 0; j < NN; ++j) {
      float v = d2[l][j];
      if (v < bv[KNN - 1]) {
        int p = KNN - 1;
        while (p > 0 && v < bv[p - 1]) { bv[p] = bv[p - 1]; bi[p] = bi[p - 1]; --p; }
        bv[p] = v; bi[p] = j;
      }
    }
    int i = rowbase + l;
    #pragma unroll
    for (int s = 0; s < KNN; ++s) {
      int j = bi[s];
      atomicOr(&adj[(b * NN + i) * ADJW + (j >> 5)], 1u << (j & 31));
      atomicOr(&adj[(b * NN + j) * ADJW + (i >> 5)], 1u << (i & 31));  // symmetrize
    }
  }
}

// ---------------- kernel 4: bitmask mean aggregation ----------------
// Wave per node; lane handles 4 features (float4). Writes bf16 into Acat left half.
__global__ void __launch_bounds__(256)
sage_aggregate_kernel(const float* __restrict__ h, const unsigned* __restrict__ adj,
                      unsigned* __restrict__ AcatDst) {
  int node = blockIdx.x * 8 + (threadIdx.x >> 5);
  int lane = threadIdx.x & 31;
  int i = node % NN;
  int base = node - i;               // b*NN
  float a0 = 0.f, a1 = 0.f, a2 = 0.f, a3 = 0.f;
  int deg = 0;
  const unsigned* row = adj + node * ADJW;
  for (int w = 0; w < 13; ++w) {
    unsigned bits = row[w];
    deg += __popc(bits);
    while (bits) {
      int t = __ffs(bits) - 1;
      bits &= bits - 1;
      const float4 v = *(const float4*)(h + (size_t)(base + w * 32 + t) * FF + lane * 4);
      a0 += v.x; a1 += v.y; a2 += v.z; a3 += v.w;
    }
  }
  float inv = 1.f / (float)(deg > 0 ? deg : 1);
  a0 *= inv; a1 *= inv; a2 *= inv; a3 *= inv;
  unsigned lo = f2bf(a0) | (f2bf(a1) << 16);
  unsigned hi = f2bf(a2) | (f2bf(a3) << 16);
  AcatDst[node * 128 + lane * 2 + 0] = lo;
  AcatDst[node * 128 + lane * 2 + 1] = hi;
}

// ---------------- kernel 5: fused SAGE GEMM (WMMA bf16 + TDM A-tile staging) ----------------
// Block = 256 (8 waves): 32 rows x 128 cols. Each wave: one 16-col tile, two row halves.
// A-tile (32 rows x 512B = 16KB contiguous) staged into LDS by the Tensor Data Mover.
__global__ void __launch_bounds__(256)
sage_gemm_kernel(const unsigned* __restrict__ Acat, const unsigned short* __restrict__ WcatF,
                 const float* __restrict__ bias, float* __restrict__ out_f32,
                 unsigned short* __restrict__ out_bf16 /* Acat_next ushort view or null */,
                 int relu) {
  __shared__ unsigned At[32][128];   // 32 rows x 256 bf16 (16 KB)
  int node0 = blockIdx.x * 32;
  int tid = threadIdx.x;

#if defined(__HIP_DEVICE_COMPILE__) && __has_builtin(__builtin_amdgcn_tensor_load_to_lds) && __has_builtin(__builtin_amdgcn_s_wait_tensorcnt)
  if (tid < 32) {
    // Build 1-D TDM descriptor: copy 2048 x 8B from Acat+node0*512 -> LDS At.
    unsigned ldsOff = (unsigned)(unsigned long long)(void*)&At[0][0];   // LDS aperture low bits
    unsigned long long ga = (unsigned long long)(const void*)(Acat + (size_t)node0 * 128);
    v4u g0 = { 1u,                                    // count=1 (valid descriptor)
               ldsOff,                                // lds_addr [63:32]
               (unsigned)(ga & 0xffffffffu),          // global_addr [95:64]
               (unsigned)((ga >> 32) & 0x01ffffffu) | 0x80000000u };  // addr[56:32] | type=2
    v8i_ g1 = { (int)(3u << 16),                      // data_size=3 (8B), wg_mask=0
                (int)(2048u << 16),                   // tensor_dim0[15:0]=2048 at [63:48]
                (int)(1u << 16),                      // tensor_dim1=1 at [95:80]
                (int)(2048u << 16),                   // tile_dim0=2048 at [127:112]
                0,                                    // tile_dim1=0, tile_dim2=0
                2048,                                 // tensor_dim0_stride[31:0]
                0, 0 };
    v4i_ gz = { 0, 0, 0, 0 };
#if __clang_major__ >= 23
    v8i_ gz8 = { 0, 0, 0, 0, 0, 0, 0, 0 };
    __builtin_amdgcn_tensor_load_to_lds(g0, g1, gz, gz, gz8, 0);
#else
    __builtin_amdgcn_tensor_load_to_lds(g0, g1, gz, gz, 0);
#endif
    __builtin_amdgcn_s_wait_tensorcnt(0);
  }
#else
  for (int idx = tid; idx < 32 * 128; idx += 256) {
    int r = idx >> 7, c = idx & 127;
    At[r][c] = Acat[(node0 + r) * 128 + c];
  }
#endif
  __syncthreads();

  int wv = tid >> 5, l = tid & 31;
  int half = l >> 4, lm = l & 15;
  int ncol = wv * 16 + lm;

  v8f acc0 = {};
  v8f acc1 = {};
  #pragma unroll
  for (int kt = 0; kt < 8; ++kt) {
    int k0 = kt * 32;
    V16 a0, a1, bm;
    // B fragment: 32 contiguous bytes per lane from fragment-ready weights
    const uint4* wp = (const uint4*)((const unsigned*)WcatF + (size_t)(((wv * 8 + kt) * 32 + l) * 8));
    *(uint4*)&bm.d[0] = wp[0];
    *(uint4*)&bm.d[4] = wp[1];
    #pragma unroll
    for (int e = 0; e < 16; ++e) {
      int kla = ((e >> 3) << 4) + (half << 3) + (e & 7);
      a0.u[e] = lds_u16(At[lm],      k0 + kla);
      a1.u[e] = lds_u16(At[16 + lm], k0 + kla);
    }
    acc0 = __builtin_amdgcn_wmma_f32_16x16x32_bf16(false, a0.v, false, bm.v,
                                                   (short)0, acc0, false, false);
    acc1 = __builtin_amdgcn_wmma_f32_16x16x32_bf16(false, a1.v, false, bm.v,
                                                   (short)0, acc1, false, false);
  }

  float bv = bias[ncol];
  #pragma unroll
  for (int r = 0; r < 8; ++r) {
    int row = r + (half << 3);
    float v0 = acc0[r] + bv;
    float v1 = acc1[r] + bv;
    if (relu) { v0 = v0 > 0.f ? v0 : 0.f; v1 = v1 > 0.f ? v1 : 0.f; }
    int nA = node0 + row;
    int nB = node0 + 16 + row;
    out_f32[(size_t)nA * FF + ncol] = v0;
    out_f32[(size_t)nB * FF + ncol] = v1;
    if (out_bf16) {
      out_bf16[(size_t)nA * 256 + 128 + ncol] = (unsigned short)f2bf(v0);
      out_bf16[(size_t)nB * 256 + 128 + ncol] = (unsigned short)f2bf(v1);
    }
  }
}

// ---------------- launch ----------------
extern "C" void kernel_launch(void* const* d_in, const int* in_sizes, int n_in,
                              void* d_out, int out_size, void* d_ws, size_t ws_size,
                              hipStream_t stream) {
  (void)in_sizes; (void)n_in; (void)out_size; (void)ws_size;
  const float* x   = (const float*)d_in[0];
  const float* W1l = (const float*)d_in[1];
  const float* b1l = (const float*)d_in[2];
  const float* W1r = (const float*)d_in[3];
  const float* W2l = (const float*)d_in[4];
  const float* b2l = (const float*)d_in[5];
  const float* W2r = (const float*)d_in[6];
  float* out = (float*)d_out;

  char* ws = (char*)d_ws;
  // ws layout (bytes)
  unsigned*       adj   = (unsigned*)(ws + 0);               //  3,276,800
  unsigned*       Acat1 = (unsigned*)(ws + 3276800);         // 26,214,400 (51200 x 256 bf16)
  unsigned*       Acat2 = (unsigned*)(ws + 29491200);        // 26,214,400
  float*          h1    = (float*)   (ws + 55705600);        // 26,214,400
  unsigned short* WcatF1 = (unsigned short*)(ws + 81920000); //     65,536
  unsigned short* WcatF2 = (unsigned short*)(ws + 81985536); //     65,536

  zero_kernel<<<(NODES * ADJW + 255) / 256, 256, 0, stream>>>(adj, NODES * ADJW);
  cvt_x_kernel<<<(NODES * 64 + 255) / 256, 256, 0, stream>>>(x, Acat1);
  prep_w_kernel<<<256, 256, 0, stream>>>(W1l, W1r, W2l, W2r, WcatF1, WcatF2);

  gram_topk_kernel<<<BB * 25, 32, 0, stream>>>(Acat1, adj);

  // layer 1
  sage_aggregate_kernel<<<NODES / 8, 256, 0, stream>>>(x, adj, Acat1);
  sage_gemm_kernel<<<NODES / 32, 256, 0, stream>>>(Acat1, WcatF1, b1l, h1,
                                                   (unsigned short*)Acat2, 1);
  // layer 2
  sage_aggregate_kernel<<<NODES / 8, 256, 0, stream>>>(h1, adj, Acat2);
  sage_gemm_kernel<<<NODES / 32, 256, 0, stream>>>(Acat2, WcatF2, b2l, out,
                                                   (unsigned short*)nullptr, 0);
}
